// STETernaryLinear_21457656611094
// MI455X (gfx1250) — matrix operand order repaired and checked by reading, not compiled, not run on hardware
//
#include <hip/hip_runtime.h>

typedef __attribute__((ext_vector_type(16))) _Float16 v16h;
typedef __attribute__((ext_vector_type(8)))  _Float16 v8h;
typedef __attribute__((ext_vector_type(8)))  float    v8f;

#define M_DIM 8192
#define N_DIM 4096
#define K_DIM 4096
#define W_ELEMS (4096 * 4096)          // 16,777,216
#define X_ELEMS (8192 * 4096)          // 33,554,432
#define PF_DIST 512                    // halves = 1024 B = 16 K-steps ahead
#define ROW16_B 131072                 // 16 rows * 4096 halves * 2 B

// ---------------- workspace layout ----------------
// [0, 4)            : float threshold
// [256, 256+16KB)   : 4096 float partial sums
// [32768, +32MB)    : f16 ternary weights  [4096][4096]
// [32768+32MB,+64MB): f16 x               [8192][4096]
#define PARTIALS_OFF 256
#define WH_OFF       32768
#define XH_OFF       (32768 + (size_t)W_ELEMS * 2)

// ---- 1) per-block partial sums of |w| (deterministic, no float atomics) ----
__global__ __launch_bounds__(256) void k_abs_partial(const float* __restrict__ w,
                                                     float* __restrict__ partials) {
  __shared__ float sm[256];
  size_t base = (size_t)blockIdx.x * 4096;
  float s = 0.0f;
  for (int i = threadIdx.x; i < 4096; i += 256) s += fabsf(w[base + i]);
  sm[threadIdx.x] = s;
  __syncthreads();
  for (int off = 128; off > 0; off >>= 1) {
    if (threadIdx.x < off) sm[threadIdx.x] += sm[threadIdx.x + off];
    __syncthreads();
  }
  if (threadIdx.x == 0) partials[blockIdx.x] = sm[0];
}

// ---- 2) final reduction -> threshold = 0.05 * mean(|w|) ----
__global__ __launch_bounds__(256) void k_thr(const float* __restrict__ partials,
                                             float* __restrict__ thr_out) {
  __shared__ float sm[256];
  float s = 0.0f;
  for (int i = threadIdx.x; i < 4096; i += 256) s += partials[i];
  sm[threadIdx.x] = s;
  __syncthreads();
  for (int off = 128; off > 0; off >>= 1) {
    if (threadIdx.x < off) sm[threadIdx.x] += sm[threadIdx.x + off];
    __syncthreads();
  }
  if (threadIdx.x == 0) thr_out[0] = 0.05f * sm[0] / (float)W_ELEMS;
}

// ---- 3) ternarize weights into f16 ----
__global__ __launch_bounds__(256) void k_ternarize(const float* __restrict__ w,
                                                   const float* __restrict__ thr_p,
                                                   _Float16* __restrict__ wh, int n) {
  const float thr = thr_p[0];
  int stride = gridDim.x * blockDim.x;
  for (int i = blockIdx.x * blockDim.x + threadIdx.x; i < n; i += stride) {
    float v = w[i];
    float q = (v > thr) ? 1.0f : ((v < -thr) ? -1.0f : 0.0f);
    wh[i] = (_Float16)q;
  }
}

// ---- 4) convert activations to f16 ----
__global__ __launch_bounds__(256) void k_cvt_x(const float* __restrict__ x,
                                               _Float16* __restrict__ xh, int n) {
  int stride = gridDim.x * blockDim.x;
  for (int i = blockIdx.x * blockDim.x + threadIdx.x; i < n; i += stride)
    xh[i] = (_Float16)x[i];
}

// ---- WGP-scope prefetch (default scope 0 = WGP -> pulls line into WGP$) ----
// __builtin_prefetch only reaches SE/SYS scopes which stop at GL2 (where the
// data already resides). Row displacement + 1KB prefetch distance live in the
// 24-bit immediate so the compiler reuses the loads' base VGPR pairs.
template <int BYTE_OFF>
__device__ inline void prefetch_wgp(const _Float16* p) {
  asm volatile("global_prefetch_b8 %0, off offset:%c1" ::"v"(p),
               "i"(BYTE_OFF + 1024));
}

// ---- A-fragment load: 16-bit A 16x32, lane holds contiguous K [c*16,+16) ----
__device__ inline v16h load_a_frag(const _Float16* p) {
  v8h lo = *(const v8h*)(p);        // K = c*16 .. c*16+7
  v8h hi = *(const v8h*)(p + 8);    // K = c*16+8 .. c*16+15
  return __builtin_shufflevector(lo, hi, 0, 1, 2, 3, 4, 5, 6, 7,
                                         8, 9, 10, 11, 12, 13, 14, 15);
}

__device__ inline v8f wmma_f16(v16h a, v16h b, v8f c) {
  return __builtin_amdgcn_wmma_f32_16x16x32_f16(false, a, false, b,
                                                (short)0, c, false, false);
}

// ---- 5) WMMA GEMM: out[m,n] = sum_k xh[m,k] * wh[n,k] ----
// block = 256 threads = 8 waves (4 rows x 2 cols); wave tile = 32x64;
// block tile = 128x128. 8 WMMAs per K-step, 12 b128 loads (1.5 loads/WMMA);
// one 6-stream WGP prefetch per 128B (= per two K-steps = per L2 line).
__global__ __launch_bounds__(256) void k_gemm(const _Float16* __restrict__ XH,
                                              const _Float16* __restrict__ WH,
                                              float* __restrict__ out) {
  const int lane = threadIdx.x & 31;
  const int wave = threadIdx.x >> 5;
  const int l    = lane & 15;   // row of A / col of B / N index of C
  const int half = lane >> 4;   // K-chunk selector (A,B); M+8 selector (C)
  const int wr = wave & 3;      // 0..3
  const int wc = wave >> 2;     // 0..1
  const int m_w = blockIdx.y * 128 + wr * 32;
  const int n_w = blockIdx.x * 128 + wc * 64;

  // A lane base: row (m_w + i*16 + l), contiguous 16 elems at K half*16
  const _Float16* pa0 = XH + (size_t)(m_w + l) * K_DIM + half * 16;
  const _Float16* pa1 = pa0 + (size_t)16 * K_DIM;
  // B lane base: col (n_w + j*16 + l), contiguous 16 elems at K half*16
  const _Float16* pb0 = WH + (size_t)(n_w + l) * K_DIM + half * 16;
  const _Float16* pb1 = pb0 + (size_t)16 * K_DIM;
  const _Float16* pb2 = pb0 + (size_t)32 * K_DIM;
  const _Float16* pb3 = pb0 + (size_t)48 * K_DIM;

  v8f c[2][4];
#pragma unroll
  for (int i = 0; i < 2; ++i)
#pragma unroll
    for (int j = 0; j < 4; ++j) c[i][j] = (v8f){};

#define GEMM_STEP(KB)                                                          \
  {                                                                            \
    v16h a[2], b[4];                                                           \
    a[0] = load_a_frag(pa0 + (KB));                                            \
    a[1] = load_a_frag(pa1 + (KB));                                            \
    b[0] = *(const v16h*)(pb0 + (KB));                                         \
    b[1] = *(const v16h*)(pb1 + (KB));                                         \
    b[2] = *(const v16h*)(pb2 + (KB));                                         \
    b[3] = *(const v16h*)(pb3 + (KB));                                         \
    _Pragma("unroll") for (int i = 0; i < 2; ++i)                              \
        _Pragma("unroll") for (int j = 0; j < 4; ++j)                          \
            c[i][j] = wmma_f16(a[i], b[j], c[i][j]);                           \
  }

  int kb = 0;
  // main loop: two K-steps (128 B) per body, one line-prefetch per stream
  for (; kb < K_DIM - PF_DIST; kb += 64) {
    prefetch_wgp<0>(pa0 + kb);
    prefetch_wgp<ROW16_B>(pa0 + kb);
    prefetch_wgp<0>(pb0 + kb);
    prefetch_wgp<ROW16_B>(pb0 + kb);
    prefetch_wgp<2 * ROW16_B>(pb0 + kb);
    prefetch_wgp<3 * ROW16_B>(pb0 + kb);
    GEMM_STEP(kb)
    GEMM_STEP(kb + 32)
  }
  // tail: last PF_DIST halves, already prefetched
#pragma unroll 2
  for (; kb < K_DIM; kb += 32) GEMM_STEP(kb)
#undef GEMM_STEP

  // C/D layout: VGPR r, lanes 0-15 -> (M=r, N=l); lanes 16-31 -> (M=r+8, N=l)
#pragma unroll
  for (int i = 0; i < 2; ++i)
#pragma unroll
    for (int j = 0; j < 4; ++j) {
      const int gn = n_w + j * 16 + l;
#pragma unroll
      for (int r = 0; r < 8; ++r) {
        const int gm = m_w + i * 16 + r + half * 8;
        out[(size_t)gm * N_DIM + gn] = c[i][j][r];
      }
    }
}

extern "C" void kernel_launch(void* const* d_in, const int* in_sizes, int n_in,
                              void* d_out, int out_size, void* d_ws, size_t ws_size,
                              hipStream_t stream) {
  const float* x = (const float*)d_in[0];   // [8192,4096]
  const float* w = (const float*)d_in[1];   // [4096,4096]
  float* out = (float*)d_out;               // [8192,4096]

  char* ws = (char*)d_ws;
  float*    thr_p    = (float*)ws;
  float*    partials = (float*)(ws + PARTIALS_OFF);
  _Float16* wh       = (_Float16*)(ws + WH_OFF);
  _Float16* xh       = (_Float16*)(ws + XH_OFF);

  k_abs_partial<<<4096, 256, 0, stream>>>(w, partials);
  k_thr<<<1, 256, 0, stream>>>(partials, thr_p);
  k_ternarize<<<4096, 256, 0, stream>>>(w, thr_p, wh, W_ELEMS);
  k_cvt_x<<<8192, 256, 0, stream>>>(x, xh, X_ELEMS);

  dim3 grid(N_DIM / 128, M_DIM / 128);   // (32, 64)
  k_gemm<<<grid, 256, 0, stream>>>(xh, wh, out);
}
